// VQAttention_20375324852789
// MI455X (gfx1250) — matrix operand order repaired and checked by reading, not compiled, not run on hardware
//
#include <hip/hip_runtime.h>
#include <hip/hip_bf16.h>
#include <stdint.h>

#define B_    2
#define L_    2048
#define D_    1024
#define H_    16
#define HD_   64
#define NCODE_ 512
#define M_    (B_ * L_)
#define OUT_ELEMS ((size_t)B_ * L_ * D_)   // 4194304
#define LC_OFF   (OUT_ELEMS)
#define LCB_OFF  (OUT_ELEMS + 1)
#define SC_OFF   (OUT_ELEMS + 2)

typedef __attribute__((ext_vector_type(16))) __bf16 v16bf;
typedef __attribute__((ext_vector_type(8)))  __bf16 v8bf;
typedef __attribute__((ext_vector_type(8)))  float  v8f;
typedef __attribute__((ext_vector_type(4)))  int    v4i;

// f32 -> bf16 round-to-nearest-even, raw u16
__device__ __forceinline__ unsigned short f2bf(float f) {
    union { float f; uint32_t u; } v; v.f = f;
    uint32_t u = v.u;
    uint32_t r = (u + 0x7FFFu + ((u >> 16) & 1u)) >> 16;
    return (unsigned short)r;
}

__device__ __forceinline__ void sched_fence() {
#if __has_builtin(__builtin_amdgcn_sched_barrier)
    __builtin_amdgcn_sched_barrier(0);
#endif
}

// ---- CDNA5 async global->LDS copy (16 bytes per lane), ASYNCcnt tracked ----
__device__ __forceinline__ void async_cp_b128(const unsigned short* g, unsigned short* l) {
#if defined(__gfx1250__) && __has_builtin(__builtin_amdgcn_global_load_async_to_lds_b128)
    unsigned short* gnc = const_cast<unsigned short*>(g);
    __builtin_amdgcn_global_load_async_to_lds_b128(
        (__attribute__((address_space(1))) v4i*)gnc,
        (__attribute__((address_space(3))) v4i*)l, 0, 0);
#else
    unsigned lofs = (unsigned)(unsigned long)
        (__attribute__((address_space(3))) unsigned short*)l;
    asm volatile("global_load_async_to_lds_b128 %0, %1, off"
                 :: "v"(lofs), "v"((unsigned long long)(const void*)g)
                 : "memory");
#endif
}

__device__ __forceinline__ void wait_async0() {
#if defined(__gfx1250__) && __has_builtin(__builtin_amdgcn_s_wait_asynccnt)
    __builtin_amdgcn_s_wait_asynccnt(0);
#else
    asm volatile("s_wait_asynccnt 0x0" ::: "memory");
#endif
}

// A-fragment load (contiguous k order, matched by the B loads): lane<16 -> K {0..15}; lane>=16 -> K {16..31}
__device__ __forceinline__ v16bf ldA(const unsigned short* rowp, int kb) {
    v16bf a;
    ((v8bf*)&a)[0] = *(const v8bf*)(rowp + kb);
    ((v8bf*)&a)[1] = *(const v8bf*)(rowp + kb + 8);
    return a;
}

__device__ __forceinline__ v8f wmma_bf16(v16bf a, v16bf b, v8f c) {
    return __builtin_amdgcn_wmma_f32_16x16x32_bf16(false, a, false, b, (short)0, c, false, false);
}

// ---------------------------------------------------------------- init
__global__ void init_out_kernel(float* out) {
    if (threadIdx.x < 2) out[LC_OFF + threadIdx.x] = 0.0f;
}

// ---------------------------------------------------------------- f32 -> bf16
__global__ void cvt_f32_bf16_kernel(const float* __restrict__ in,
                                    unsigned short* __restrict__ out, int n) {
    int i = blockIdx.x * 256 + threadIdx.x;
    if (i < n) out[i] = f2bf(in[i]);
}

// ---------------------------------------------------------------- WMMA GEMM
// C = A[M,K] * W[N,K]^T, bf16 in, optional f32 and/or bf16 out.
// 128x128 tile, K-chunk 32, double-buffered LDS fed by async global->LDS DMA.
__device__ __forceinline__ void gemm_issue_tile(const unsigned short* __restrict__ A,
                                                const unsigned short* __restrict__ W,
                                                unsigned short* As, unsigned short* Ws,
                                                int mt, int nt, int kk, int K, int tid) {
    // each tile: 128 rows x 32 halves = 512 chunks of 16B
    for (int i = tid; i < 512; i += 256) {
        int r = i >> 2, c8 = (i & 3) * 8;
        async_cp_b128(&A[(size_t)(mt + r) * K + kk + c8], &As[r * 32 + c8]);
        async_cp_b128(&W[(size_t)(nt + r) * K + kk + c8], &Ws[r * 32 + c8]);
    }
}

__global__ __launch_bounds__(256)
void gemm_bf16_kernel(const unsigned short* __restrict__ A,
                      const unsigned short* __restrict__ W,
                      float* __restrict__ Cf, unsigned short* __restrict__ Cb,
                      int M, int N, int K) {
    __shared__ __align__(64) unsigned short As[2][128 * 32];
    __shared__ __align__(64) unsigned short Ws[2][128 * 32];
    const int tid = threadIdx.x, lane = tid & 31, w = tid >> 5;
    const int mt = blockIdx.y * 128, nt = blockIdx.x * 128;
    const int cc = lane & 15;
    const int ak = (lane & 16) ? 16 : 0;
    const int bk = (lane & 16) ? 16 : 0;

    v8f acc[8];
#pragma unroll
    for (int j = 0; j < 8; j++) acc[j] = (v8f){0.f,0.f,0.f,0.f,0.f,0.f,0.f,0.f};

    gemm_issue_tile(A, W, As[0], Ws[0], mt, nt, 0, K, tid);
    wait_async0();
    __syncthreads();

    const int nk = K >> 5;
    for (int kt = 0; kt < nk; kt++) {
        const int cur = kt & 1;
        if (kt + 1 < nk)
            gemm_issue_tile(A, W, As[1 - cur], Ws[1 - cur], mt, nt, (kt + 1) << 5, K, tid);

        // fragment-load clause ...
        v16bf af = ldA(&As[cur][(w * 16 + cc) * 32], ak);
        v16bf bfr[8];
#pragma unroll
        for (int j = 0; j < 8; j++)
            bfr[j] = *(const v16bf*)&Ws[cur][(j * 16 + cc) * 32 + bk];
        sched_fence();
        // ... then back-to-back WMMA burst
#pragma unroll
        for (int j = 0; j < 8; j++)
            acc[j] = wmma_bf16(af, bfr[j], acc[j]);
        sched_fence();

        wait_async0();     // next buffer landed
        __syncthreads();   // everyone done reading cur before it is overwritten
    }

    const int rbase = mt + w * 16 + ((lane & 16) ? 8 : 0);
    if (Cf) {
#pragma unroll
        for (int j = 0; j < 8; j++)
#pragma unroll
            for (int e = 0; e < 8; e++)
                Cf[(size_t)(rbase + e) * N + nt + j * 16 + cc] = acc[j][e];
    }
    if (Cb) {
#pragma unroll
        for (int j = 0; j < 8; j++)
#pragma unroll
            for (int e = 0; e < 8; e++)
                Cb[(size_t)(rbase + e) * N + nt + j * 16 + cc] = f2bf(acc[j][e]);
    }
}

// ---------------------------------------------------------------- VQ argmin
__global__ __launch_bounds__(256)
void vq_kernel(const float* __restrict__ Kf, const float* __restrict__ cb,
               unsigned short* __restrict__ khat, float* __restrict__ out) {
    __shared__ float kb[8][64];
    const int lane = threadIdx.x & 31, w = threadIdx.x >> 5;
    const int flat = blockIdx.x * 8 + w;          // b*H*L + h*L + l
    const int l = flat & (L_ - 1);
    const int bh = flat >> 11;
    const int h = bh & (H_ - 1), b = bh >> 4;

    const float* kp = Kf + ((size_t)(b * L_ + l)) * D_ + h * HD_;
    kb[w][lane] = kp[lane];
    kb[w][lane + 32] = kp[lane + 32];
    asm volatile("s_wait_dscnt 0x0" ::: "memory");

    float k2 = 0.f;
#pragma unroll
    for (int d = 0; d < 64; d++) { float kv = kb[w][d]; k2 = __fmaf_rn(kv, kv, k2); }

    const float* cbh = cb + (size_t)h * NCODE_ * HD_;
    float best = 3.0e38f; int bidx = NCODE_;
    for (int ci = 0; ci < NCODE_ / 32; ci++) {
        int c = lane + ci * 32;
        const float* cp = cbh + (size_t)c * HD_;
        float acc = 0.f;
#pragma unroll
        for (int d = 0; d < 64; d++) {
            float cv = cp[d];
            acc = __fmaf_rn(cv, cv - 2.0f * kb[w][d], acc);
        }
        if (acc < best) { best = acc; bidx = c; }
    }
    for (int m = 16; m >= 1; m >>= 1) {
        float ob = __shfl_xor(best, m, 32);
        int   oi = __shfl_xor(bidx, m, 32);
        if (ob < best || (ob == best && oi < bidx)) { best = ob; bidx = oi; }
    }
    if (lane == 0) {
        out[SC_OFF + flat] = (float)bidx;
        float v = (k2 + best) * (1.0f / (float)(B_ * H_ * L_));
        atomicAdd(&out[LC_OFF], v);
        atomicAdd(&out[LCB_OFF], v);
    }
    int code = __shfl(bidx, 0, 32);
    const float* cp = cbh + (size_t)code * HD_;
    unsigned short* kh = khat + ((size_t)(b * L_ + l)) * D_ + h * HD_;
    kh[lane]      = f2bf(cp[lane]);
    kh[lane + 32] = f2bf(cp[lane + 32]);
}

// ---------------------------------------------------------------- flash attention
// bf16 in (Q, k_hat, V), bf16 out. Q + K tiles via async global->LDS DMA.
__global__ __launch_bounds__(256)
void attn_kernel(const unsigned short* __restrict__ Qb,
                 const unsigned short* __restrict__ Khb,
                 const unsigned short* __restrict__ Vb,
                 unsigned short* __restrict__ attnb) {
    __shared__ __align__(64) unsigned short Qs[128 * 64];
    __shared__ __align__(64) unsigned short Ks[32 * 64];
    __shared__ __align__(64) unsigned short Vt[64 * 32];   // [d][t]
    __shared__ __align__(64) unsigned short Ps[8 * 16 * 32];
    const int tid = threadIdx.x, lane = tid & 31, w = tid >> 5;
    const int qt = blockIdx.x, bh = blockIdx.y;
    const int h = bh & (H_ - 1), b = bh >> 4;
    const size_t base = ((size_t)b * L_) * D_ + (size_t)h * HD_;
    const int cc = lane & 15;
    const int hh = (lane & 16) ? 8 : 0;
    const int ak = (lane & 16) ? 16 : 0;
    const int bk = (lane & 16) ? 16 : 0;
    const int rglob0 = qt * 128 + w * 16 + hh;
    const float scale = 0.125f;   // HD^-0.5

    // Q tile: 128 rows x 64 halves = 1024 chunks of 16B, async DMA
#pragma unroll
    for (int j = 0; j < 4; j++) {
        int idx = tid + j * 256;
        int r = idx >> 3, c8 = (idx & 7) * 8;
        async_cp_b128(&Qb[base + (size_t)(qt * 128 + r) * D_ + c8], &Qs[r * 64 + c8]);
    }
    float mrow[8], lrow[8];
    v8f o[4];
#pragma unroll
    for (int e = 0; e < 8; e++) { mrow[e] = -1.0e30f; lrow[e] = 0.f; }
#pragma unroll
    for (int dc = 0; dc < 4; dc++) o[dc] = (v8f){0.f,0.f,0.f,0.f,0.f,0.f,0.f,0.f};
    wait_async0();
    __syncthreads();

    const int nT = (qt + 1) * 4;
    for (int tt = 0; tt < nT; tt++) {
        {   // K tile: 32 x 64 halves = 256 chunks, one async per thread
            int r = tid >> 3, c8 = (tid & 7) * 8;
            async_cp_b128(&Khb[base + (size_t)(tt * 32 + r) * D_ + c8], &Ks[r * 64 + c8]);
        }
        for (int i = tid; i < 32 * 64; i += 256) {   // V tile, transposed store
            int r = i >> 6, d = i & 63;
            Vt[d * 32 + r] = Vb[base + (size_t)(tt * 32 + r) * D_ + d];
        }
        wait_async0();
        __syncthreads();

        v8f s0 = (v8f){0.f,0.f,0.f,0.f,0.f,0.f,0.f,0.f};
        v8f s1 = (v8f){0.f,0.f,0.f,0.f,0.f,0.f,0.f,0.f};
        const unsigned short* qrow = &Qs[(w * 16 + cc) * 64];
        {   // fragment-load clause, then 4 WMMAs back-to-back
            v16bf aq0 = ldA(qrow, ak);
            v16bf aq1 = ldA(qrow + 32, ak);
            v16bf b00 = *(const v16bf*)&Ks[cc * 64 + bk];
            v16bf b01 = *(const v16bf*)&Ks[(16 + cc) * 64 + bk];
            v16bf b10 = *(const v16bf*)&Ks[cc * 64 + 32 + bk];
            v16bf b11 = *(const v16bf*)&Ks[(16 + cc) * 64 + 32 + bk];
            sched_fence();
            s0 = wmma_bf16(aq0, b00, s0);
            s1 = wmma_bf16(aq0, b01, s1);
            s0 = wmma_bf16(aq1, b10, s0);
            s1 = wmma_bf16(aq1, b11, s1);
            sched_fence();
        }

        const int col0 = tt * 32 + cc, col1 = col0 + 16;
#pragma unroll
        for (int e = 0; e < 8; e++) {
            int row = rglob0 + e;
            float a  = (col0 <= row) ? s0[e] * scale : -2.0e30f;
            float bb = (col1 <= row) ? s1[e] * scale : -2.0e30f;
            float rm = fmaxf(a, bb);
            for (int mm = 8; mm >= 1; mm >>= 1)
                rm = fmaxf(rm, __shfl_xor(rm, mm, 32));
            float mn = fmaxf(mrow[e], rm);
            float p0 = __expf(a - mn), p1 = __expf(bb - mn);
            float rs = p0 + p1;
            for (int mm = 8; mm >= 1; mm >>= 1)
                rs += __shfl_xor(rs, mm, 32);
            float sf = __expf(mrow[e] - mn);
            lrow[e] = lrow[e] * sf + rs;
            mrow[e] = mn;
#pragma unroll
            for (int dc = 0; dc < 4; dc++) o[dc][e] *= sf;
            Ps[w * 512 + (e + hh) * 32 + cc]      = f2bf(p0);
            Ps[w * 512 + (e + hh) * 32 + 16 + cc] = f2bf(p1);
        }
        asm volatile("s_wait_dscnt 0x0" ::: "memory");

        {   // P*V: fragment-load clause, then 4 WMMAs back-to-back
            v16bf ap = ldA(&Ps[w * 512 + cc * 32], ak);
            v16bf bv[4];
#pragma unroll
            for (int dc = 0; dc < 4; dc++)
                bv[dc] = *(const v16bf*)&Vt[(dc * 16 + cc) * 32 + bk];
            sched_fence();
#pragma unroll
            for (int dc = 0; dc < 4; dc++)
                o[dc] = wmma_bf16(ap, bv[dc], o[dc]);
            sched_fence();
        }
        __syncthreads();
    }

#pragma unroll
    for (int e = 0; e < 8; e++) {
        int row = rglob0 + e;
        float inv = 1.0f / lrow[e];
#pragma unroll
        for (int dc = 0; dc < 4; dc++)
            attnb[base + (size_t)row * D_ + dc * 16 + cc] = f2bf(o[dc][e] * inv);
    }
}

// ---------------------------------------------------------------- launch
extern "C" void kernel_launch(void* const* d_in, const int* in_sizes, int n_in,
                              void* d_out, int out_size, void* d_ws, size_t ws_size,
                              hipStream_t stream) {
    const float* x  = (const float*)d_in[0];
    const float* Wq = (const float*)d_in[1];
    const float* Wk = (const float*)d_in[2];
    const float* Wv = (const float*)d_in[3];
    const float* Wo = (const float*)d_in[4];
    const float* cb = (const float*)d_in[5];
    float* out = (float*)d_out;

    char* ws = (char*)d_ws;
    size_t off = 0;
    auto alloc = [&](size_t bytes) -> void* {
        void* p = ws + off;
        off += (bytes + 255) & ~(size_t)255;
        return p;
    };
    unsigned short* xb    = (unsigned short*)alloc((size_t)M_ * D_ * 2);
    unsigned short* Wqb   = (unsigned short*)alloc((size_t)D_ * D_ * 2);
    unsigned short* Wkb   = (unsigned short*)alloc((size_t)D_ * D_ * 2);
    unsigned short* Wvb   = (unsigned short*)alloc((size_t)D_ * D_ * 2);
    unsigned short* Wob   = (unsigned short*)alloc((size_t)D_ * D_ * 2);
    unsigned short* Qb    = (unsigned short*)alloc((size_t)M_ * D_ * 2);
    unsigned short* Vb    = (unsigned short*)alloc((size_t)M_ * D_ * 2);
    unsigned short* khatb = (unsigned short*)alloc((size_t)M_ * D_ * 2);
    unsigned short* attnb = (unsigned short*)alloc((size_t)M_ * D_ * 2);
    float* Kf = (float*)alloc((size_t)M_ * D_ * 4);

    init_out_kernel<<<1, 64, 0, stream>>>(out);

    const int nx = M_ * D_, nw = D_ * D_;
    cvt_f32_bf16_kernel<<<(nx + 255) / 256, 256, 0, stream>>>(x,  xb,  nx);
    cvt_f32_bf16_kernel<<<(nw + 255) / 256, 256, 0, stream>>>(Wq, Wqb, nw);
    cvt_f32_bf16_kernel<<<(nw + 255) / 256, 256, 0, stream>>>(Wk, Wkb, nw);
    cvt_f32_bf16_kernel<<<(nw + 255) / 256, 256, 0, stream>>>(Wv, Wvb, nw);
    cvt_f32_bf16_kernel<<<(nw + 255) / 256, 256, 0, stream>>>(Wo, Wob, nw);

    dim3 ggrid(D_ / 128, M_ / 128);
    gemm_bf16_kernel<<<ggrid, 256, 0, stream>>>(xb, Wqb, nullptr, Qb, M_, D_, D_);
    gemm_bf16_kernel<<<ggrid, 256, 0, stream>>>(xb, Wkb, Kf, nullptr, M_, D_, D_);
    gemm_bf16_kernel<<<ggrid, 256, 0, stream>>>(xb, Wvb, nullptr, Vb, M_, D_, D_);

    vq_kernel<<<(B_ * H_ * L_) / 8, 256, 0, stream>>>(Kf, cb, khatb, out);

    dim3 agrid(L_ / 128, B_ * H_);
    attn_kernel<<<agrid, 256, 0, stream>>>(Qb, khatb, Vb, attnb);

    gemm_bf16_kernel<<<ggrid, 256, 0, stream>>>(attnb, Wob, out, nullptr, M_, D_, D_);

    (void)in_sizes; (void)n_in; (void)out_size; (void)ws_size;
}
